// MultiheadAttentionWithRope_38792144617930
// MI455X (gfx1250) — compile-verified, matrix operand order checked by reading, not run on hardware
//
#include <hip/hip_runtime.h>
#include <hip/hip_bf16.h>
#include <math.h>

typedef _Float16 v16h __attribute__((ext_vector_type(16)));
typedef _Float16 v8h  __attribute__((ext_vector_type(8)));
typedef float    v8f  __attribute__((ext_vector_type(8)));

#define B_  2
#define S_  4096
#define E_  512
#define H_  8
#define D_  64
#define SCALE_ 0.125f   // 1/sqrt(64)

// ---- WMMA fragment helpers (layouts per CDNA5 ISA 7.12.2) -------------------
// A-matrix 16x32 f16: lane L holds row M=L&15; lanes 0-15 hold K {0..7,16..23},
// lanes 16-31 hold K {8..15,24..31}, packed 2 halves per VGPR in that order.
// B-matrix 32x16: identical pattern with lane = N, reading an N-major (N x K
// row-major) buffer. So one loader serves both.
__device__ __forceinline__ v16h load_frag_row(const _Float16* rowp, int lane) {
  int sel = (lane & 16) ? 8 : 0;                // halves offset: K=8.. for hi half-wave
  v8h h0 = *(const v8h*)(rowp + sel);           // K 0..7   or 8..15
  v8h h1 = *(const v8h*)(rowp + sel + 16);      // K 16..23 or 24..31
  return __builtin_shufflevector(h0, h1, 0,1,2,3,4,5,6,7,8,9,10,11,12,13,14,15);
}

__device__ __forceinline__ v8f wmma_f16(v16h a, v16h b, v8f c) {
  return __builtin_amdgcn_wmma_f32_16x16x32_f16(false, a, false, b, (short)0, c,
                                                false, false);
}

// reductions across a 16-lane half-wave (C rows live across lanes 0-15 / 16-31)
__device__ __forceinline__ float rmax16(float v) {
  v = fmaxf(v, __shfl_xor(v, 1, 32));
  v = fmaxf(v, __shfl_xor(v, 2, 32));
  v = fmaxf(v, __shfl_xor(v, 4, 32));
  v = fmaxf(v, __shfl_xor(v, 8, 32));
  return v;
}
__device__ __forceinline__ float rsum16(float v) {
  v += __shfl_xor(v, 1, 32);
  v += __shfl_xor(v, 2, 32);
  v += __shfl_xor(v, 4, 32);
  v += __shfl_xor(v, 8, 32);
  return v;
}

// ---- elementwise kernels ----------------------------------------------------
__global__ void rope_conv(const float* __restrict__ q, const float* __restrict__ k,
                          const float* __restrict__ v,
                          _Float16* qf, _Float16* kf, _Float16* vf) {
  int idx = blockIdx.x * blockDim.x + threadIdx.x;          // (b*S+s)*256 + i
  if (idx >= B_ * S_ * (E_ / 2)) return;
  int i = idx & 255;
  int bs = idx >> 8;                                        // b*S + s
  int s = bs & (S_ - 1);
  size_t off = (size_t)bs * E_ + 2 * i;
  float fr = (float)s * __powf(10000.0f, -(float)(2 * i) / (float)E_);
  float c, sn;
  __sincosf(fr, &sn, &c);
  float q1 = q[off], q2 = q[off + 1];
  qf[off]     = (_Float16)(q1 * c - q2 * sn);
  qf[off + 1] = (_Float16)(q1 * sn + q2 * c);
  float k1 = k[off], k2 = k[off + 1];
  kf[off]     = (_Float16)(k1 * c - k2 * sn);
  kf[off + 1] = (_Float16)(k1 * sn + k2 * c);
  vf[off]     = (_Float16)v[off];
  vf[off + 1] = (_Float16)v[off + 1];
}

__global__ void w_conv(const float* __restrict__ wi, const float* __restrict__ wo,
                       _Float16* wih, _Float16* woh) {
  int idx = blockIdx.x * blockDim.x + threadIdx.x;
  if (idx < 3 * E_ * E_) wih[idx] = (_Float16)wi[idx];
  if (idx < E_ * E_)     woh[idx] = (_Float16)wo[idx];
}

// ---- QKV projection: [B*S,512] x W^T -> Qh/Kh [b][h][s][d], Vt [b][h][d][s] -
__global__ void proj_gemm(const _Float16* __restrict__ qf, const _Float16* __restrict__ kf,
                          const _Float16* __restrict__ vf, const _Float16* __restrict__ wih,
                          const float* __restrict__ inb,
                          _Float16* Qh, _Float16* Kh, _Float16* Vt) {
  int lane = threadIdx.x & 31;
  int rt = blockIdx.x;               // 512 row tiles over B*S
  int j0 = blockIdx.y * 64;          // 24 strips of 64 output cols
  const _Float16* src = (j0 < 512) ? qf : (j0 < 1024 ? kf : vf);
  int r0 = rt * 16;
  const _Float16* arow = src + (size_t)(r0 + (lane & 15)) * E_;
  v8f acc[4] = {};
  for (int k0 = 0; k0 < E_; k0 += 32) {
    v16h a = load_frag_row(arow + k0, lane);
#pragma unroll
    for (int jt = 0; jt < 4; ++jt) {
      const _Float16* brow = wih + (size_t)(j0 + jt * 16 + (lane & 15)) * E_ + k0;
      acc[jt] = wmma_f16(a, load_frag_row(brow, lane), acc[jt]);
    }
  }
  int n = lane & 15, moff = (lane & 16) ? 8 : 0;
#pragma unroll
  for (int jt = 0; jt < 4; ++jt) {
    int j = j0 + jt * 16 + n;
    float bias = inb[j];
#pragma unroll
    for (int i = 0; i < 8; ++i) {
      int row = r0 + moff + i;
      int bb = row >> 12, s = row & (S_ - 1);
      float val = acc[jt][i] + bias;
      if (j0 < 512) {
        int h = j >> 6, d = j & 63;
        Qh[((size_t)(bb * H_ + h) * S_ + s) * D_ + d] = (_Float16)val;
      } else if (j0 < 1024) {
        int j2 = j - 512, h = j2 >> 6, d = j2 & 63;
        Kh[((size_t)(bb * H_ + h) * S_ + s) * D_ + d] = (_Float16)val;
      } else {
        int j2 = j - 1024, h = j2 >> 6, d = j2 & 63;
        Vt[((size_t)(bb * H_ + h) * D_ + d) * S_ + s] = (_Float16)val;   // transposed
      }
    }
  }
}

// ---- softmax pass 1: per-row running max m and sum l ------------------------
__global__ void attn_stats(const _Float16* __restrict__ Qh, const _Float16* __restrict__ Kh,
                           float* mbuf, float* lbuf) {
  int lane = threadIdx.x & 31;
  int qt = blockIdx.x, bh = blockIdx.y;
  const _Float16* qrow = Qh + ((size_t)bh * S_ + qt * 16 + (lane & 15)) * D_;
  v16h a0 = load_frag_row(qrow, lane);
  v16h a1 = load_frag_row(qrow + 32, lane);
  float m[8], l[8];
#pragma unroll
  for (int i = 0; i < 8; ++i) { m[i] = -__builtin_inff(); l[i] = 0.0f; }
  for (int kt = 0; kt < S_ / 16; ++kt) {
    const _Float16* krow = Kh + ((size_t)bh * S_ + kt * 16 + (lane & 15)) * D_;
    __builtin_prefetch(krow + 16 * D_, 0, 1);      // global_prefetch_b8: next K tile
    v8f c = {};
    c = wmma_f16(a0, load_frag_row(krow, lane), c);
    c = wmma_f16(a1, load_frag_row(krow + 32, lane), c);
#pragma unroll
    for (int i = 0; i < 8; ++i) {
      float s = c[i] * SCALE_;
      float nm = fmaxf(m[i], rmax16(s));
      float ts = rsum16(__expf(s - nm));
      l[i] = l[i] * __expf(m[i] - nm) + ts;
      m[i] = nm;
    }
  }
  if ((lane & 15) == 0) {
    int moff = (lane & 16) ? 8 : 0;
#pragma unroll
    for (int i = 0; i < 8; ++i) {
      size_t idx = (size_t)bh * S_ + qt * 16 + moff + i;
      mbuf[idx] = m[i];
      lbuf[idx] = l[i];
    }
  }
}

// ---- pass 2: probabilities -> ctx (P @ V via LDS transpose bounce) ----------
__global__ void attn_ctx(const _Float16* __restrict__ Qh, const _Float16* __restrict__ Kh,
                         const _Float16* __restrict__ Vt,
                         const float* __restrict__ mbuf, const float* __restrict__ lbuf,
                         _Float16* ctxh) {
  __shared__ __align__(16) _Float16 lds[16 * 32];
  int lane = threadIdx.x & 31;
  int qt = blockIdx.x, bh = blockIdx.y;
  int bb = bh >> 3, h = bh & 7;
  int n = lane & 15, moff = (lane & 16) ? 8 : 0;
  const _Float16* qrow = Qh + ((size_t)bh * S_ + qt * 16 + n) * D_;
  v16h a0 = load_frag_row(qrow, lane);
  v16h a1 = load_frag_row(qrow + 32, lane);
  float m[8], linv[8];
#pragma unroll
  for (int i = 0; i < 8; ++i) {
    size_t idx = (size_t)bh * S_ + qt * 16 + moff + i;
    m[i] = mbuf[idx];
    float l = lbuf[idx];
    linv[i] = (l > 0.0f) ? 1.0f / l : 0.0f;
  }
  v8f acc[4] = {};
  for (int kt2 = 0; kt2 < S_ / 32; ++kt2) {
#pragma unroll
    for (int t = 0; t < 2; ++t) {
      int key0 = kt2 * 32 + t * 16;
      const _Float16* krow = Kh + ((size_t)bh * S_ + key0 + n) * D_;
      __builtin_prefetch(krow + 32 * D_, 0, 1);    // next 32-key chunk
      v8f c = {};
      c = wmma_f16(a0, load_frag_row(krow, lane), c);
      c = wmma_f16(a1, load_frag_row(krow + 32, lane), c);
#pragma unroll
      for (int i = 0; i < 8; ++i) {
        float p = __expf(c[i] * SCALE_ - m[i]) * linv[i];
        lds[(moff + i) * 32 + t * 16 + n] = (_Float16)p;   // row-major 16x32 P tile
      }
    }
    asm volatile("s_wait_dscnt 0x0" ::: "memory");         // LDS writes -> frag reads
    v16h pa = load_frag_row(lds + n * 32, lane);           // A-frag of P (16x32 keys)
#pragma unroll
    for (int dt = 0; dt < 4; ++dt) {
      const _Float16* vrow = Vt + ((size_t)bh * D_ + dt * 16 + n) * S_ + kt2 * 32;
      acc[dt] = wmma_f16(pa, load_frag_row(vrow, lane), acc[dt]);
    }
  }
#pragma unroll
  for (int dt = 0; dt < 4; ++dt)
#pragma unroll
    for (int i = 0; i < 8; ++i) {
      int qr = qt * 16 + moff + i;
      ctxh[((size_t)(bb * S_ + qr)) * E_ + h * D_ + dt * 16 + n] = (_Float16)acc[dt][i];
    }
}

// ---- mean attention weights: recompute scores per head, plain stores --------
// One block owns a 16q x 64k tile of attn_weights[b]; loops the 8 heads in
// registers (no atomics, no zero-init pass, each element written exactly once).
__global__ void attn_wmean(const _Float16* __restrict__ Qh, const _Float16* __restrict__ Kh,
                           const float* __restrict__ mbuf, const float* __restrict__ lbuf,
                           float* __restrict__ wout) {
  int lane = threadIdx.x & 31;
  int qt = blockIdx.x;           // 256 q tiles
  int kc = blockIdx.y;           // 64 chunks of 64 keys
  int bb = blockIdx.z;           // batch
  int n = lane & 15, moff = (lane & 16) ? 8 : 0;
  float wacc[4][8] = {};
  for (int h = 0; h < H_; ++h) {
    int bh = bb * H_ + h;
    const _Float16* qrow = Qh + ((size_t)bh * S_ + qt * 16 + n) * D_;
    v16h a0 = load_frag_row(qrow, lane);
    v16h a1 = load_frag_row(qrow + 32, lane);
    float m[8], linv[8];
#pragma unroll
    for (int i = 0; i < 8; ++i) {
      size_t idx = (size_t)bh * S_ + qt * 16 + moff + i;
      m[i] = mbuf[idx];
      float l = lbuf[idx];
      linv[i] = (l > 0.0f) ? (1.0f / (H_ * l)) : 0.0f;     // fold 1/H into 1/l
    }
#pragma unroll
    for (int kt = 0; kt < 4; ++kt) {
      int key0 = kc * 64 + kt * 16;
      const _Float16* krow = Kh + ((size_t)bh * S_ + key0 + n) * D_;
      v8f c = {};
      c = wmma_f16(a0, load_frag_row(krow, lane), c);
      c = wmma_f16(a1, load_frag_row(krow + 32, lane), c);
#pragma unroll
      for (int i = 0; i < 8; ++i)
        wacc[kt][i] += __expf(c[i] * SCALE_ - m[i]) * linv[i];
    }
  }
#pragma unroll
  for (int kt = 0; kt < 4; ++kt)
#pragma unroll
    for (int i = 0; i < 8; ++i) {
      int qr = qt * 16 + moff + i;
      wout[((size_t)(bb * S_ + qr)) * S_ + kc * 64 + kt * 16 + n] = wacc[kt][i];
    }
}

// ---- output projection: ctx [B*S,512] x Wo^T + b ----------------------------
__global__ void out_gemm(const _Float16* __restrict__ ctxh, const _Float16* __restrict__ woh,
                         const float* __restrict__ ob, float* out) {
  int lane = threadIdx.x & 31;
  int r0 = blockIdx.x * 16;
  int e0 = blockIdx.y * 64;
  const _Float16* arow = ctxh + (size_t)(r0 + (lane & 15)) * E_;
  v8f acc[4] = {};
  for (int k0 = 0; k0 < E_; k0 += 32) {
    v16h a = load_frag_row(arow + k0, lane);
#pragma unroll
    for (int et = 0; et < 4; ++et) {
      const _Float16* brow = woh + (size_t)(e0 + et * 16 + (lane & 15)) * E_ + k0;
      acc[et] = wmma_f16(a, load_frag_row(brow, lane), acc[et]);
    }
  }
  int n = lane & 15, moff = (lane & 16) ? 8 : 0;
#pragma unroll
  for (int et = 0; et < 4; ++et) {
    int e = e0 + et * 16 + n;
    float bias = ob[e];
#pragma unroll
    for (int i = 0; i < 8; ++i)
      out[(size_t)(r0 + moff + i) * E_ + e] = acc[et][i] + bias;
  }
}

extern "C" void kernel_launch(void* const* d_in, const int* in_sizes, int n_in,
                              void* d_out, int out_size, void* d_ws, size_t ws_size,
                              hipStream_t stream) {
  const float* q  = (const float*)d_in[0];
  const float* k  = (const float*)d_in[1];
  const float* v  = (const float*)d_in[2];
  const float* wi = (const float*)d_in[3];
  const float* ib = (const float*)d_in[4];
  const float* wo = (const float*)d_in[5];
  const float* ob = (const float*)d_in[6];
  float* out  = (float*)d_out;                       // [B,S,E]
  float* wout = out + (size_t)B_ * S_ * E_;          // [B,S,S] mean attn weights

  const size_t NE = (size_t)B_ * S_ * E_;            // 4,194,304
  _Float16* qf   = (_Float16*)d_ws;
  _Float16* kf   = qf + NE;
  _Float16* vf   = kf + NE;
  _Float16* Qh   = vf + NE;                          // [b][h][s][d]
  _Float16* Kh   = Qh + NE;                          // [b][h][s][d]
  _Float16* Vt   = Kh + NE;                          // [b][h][d][s]
  _Float16* ctxh = Vt + NE;                          // [b][s][e]
  _Float16* wih  = ctxh + NE;                        // 3E x E
  _Float16* woh  = wih + (size_t)3 * E_ * E_;        // E x E
  float* mbuf = (float*)(woh + (size_t)E_ * E_);     // [b*h][s]
  float* lbuf = mbuf + (size_t)B_ * H_ * S_;

  rope_conv<<<(B_ * S_ * (E_ / 2) + 255) / 256, 256, 0, stream>>>(q, k, v, qf, kf, vf);
  w_conv<<<(3 * E_ * E_ + 255) / 256, 256, 0, stream>>>(wi, wo, wih, woh);
  proj_gemm<<<dim3(B_ * S_ / 16, 3 * E_ / 64), 32, 0, stream>>>(qf, kf, vf, wih, ib, Qh, Kh, Vt);
  attn_stats<<<dim3(S_ / 16, B_ * H_), 32, 0, stream>>>(Qh, Kh, mbuf, lbuf);
  attn_ctx<<<dim3(S_ / 16, B_ * H_), 32, 0, stream>>>(Qh, Kh, Vt, mbuf, lbuf, ctxh);
  attn_wmean<<<dim3(S_ / 16, S_ / 64, B_), 32, 0, stream>>>(Qh, Kh, mbuf, lbuf, wout);
  out_gemm<<<dim3(B_ * S_ / 16, E_ / 64), 32, 0, stream>>>(ctxh, woh, ob, out);
}
